// ExaoneLocalAttention_82154134438140
// MI455X (gfx1250) — matrix operand order repaired and checked by reading, not compiled, not run on hardware
//
#include <hip/hip_runtime.h>
#include <hip/hip_bf16.h>

typedef __attribute__((ext_vector_type(16))) __bf16 v16bf;
typedef __attribute__((ext_vector_type(8)))  float  v8f;
typedef __attribute__((ext_vector_type(4)))  unsigned int u32x4;
typedef __attribute__((ext_vector_type(4)))  int    i32x4;
typedef __attribute__((ext_vector_type(8)))  int    i32x8;

#define S_DIM 3072
#define E_DIM 2048
#define H_NUM 16
#define D_DIM 128
#define BL    1024
#define NB    3

#if defined(__HIP_DEVICE_COMPILE__) && __has_builtin(__builtin_amdgcn_tensor_load_to_lds) && \
    __has_builtin(__builtin_amdgcn_s_wait_tensorcnt)
#define GEMM_USE_TDM 1
#else
#define GEMM_USE_TDM 0
#endif

// ---------- bf16 helpers (bitwise, RNE) ----------
__device__ inline __bf16 f2bf(float f) {
  union { float f; unsigned u; } a; a.f = f;
  unsigned r = (a.u + 0x7fffu + ((a.u >> 16) & 1u)) >> 16;
  union { unsigned short s; __bf16 b; } c; c.s = (unsigned short)r;
  return c.b;
}
__device__ inline float bf2f(__bf16 b) {
  union { unsigned short s; __bf16 b; } c; c.b = b;
  union { unsigned u; float f; } a; a.u = ((unsigned)c.s) << 16;
  return a.f;
}

union FragCvt { struct { uint4 lo; uint4 hi; } u; v16bf v; };

// A-matrix 16x32 bf16 fragment (ISA 7.12.2): lane m = lane&15,
// kb = 0 (lanes 0-15) or 8 (lanes 16-31); VGPR0-3 hold K=kb..kb+7,
// VGPR4-7 hold K=16+kb..16+kb+7.  Row-major [m][k] source, stride in elems.
__device__ inline v16bf ldsA(const __bf16* base, int stride, int lane) {
  int m  = lane & 15;
  int kb = (lane & 16) ? 8 : 0;
  const __bf16* p = base + m * stride + kb;
  FragCvt f;
  f.u.lo = *(const uint4*)(p);
  f.u.hi = *(const uint4*)(p + 16);
  return f.v;
}
// B-matrix 32x16 bf16 fragment: lane n = lane&15, kb = 0 (lanes 0-15) or 8
// (lanes 16-31); VGPR0-3 hold K=kb..kb+7, VGPR4-7 hold K=16+kb..16+kb+7.  Source row-major [n][k].
__device__ inline v16bf ldsB(const __bf16* base, int stride, int lane) {
  int n  = lane & 15;
  int kb = (lane & 16) ? 8 : 0;
  const __bf16* p = base + n * stride + kb;
  FragCvt f;
  f.u.lo = *(const uint4*)(p);
  f.u.hi = *(const uint4*)(p + 16);
  return f.v;
}

__device__ inline v8f wmma_bf16(v16bf a, v16bf b, v8f c) {
  return __builtin_amdgcn_wmma_f32_16x16x32_bf16(false, a, false, b, (short)0, c,
                                                 false, false);
}

// low 32 bits of a canonical LDS flat address == LDS byte offset (ISA 10.2)
__device__ inline unsigned lds_byte_addr(const void* p) {
  return (unsigned)(size_t)p;
}

// ---------- f32 -> bf16 cast ----------
__global__ __launch_bounds__(256) void cast_kernel(const float* __restrict__ in,
                                                   __bf16* __restrict__ out, int n) {
  int i = (blockIdx.x * blockDim.x + threadIdx.x) * 4;
  if (i + 3 < n) {
    float4 v = *(const float4*)(in + i);
    out[i + 0] = f2bf(v.x);
    out[i + 1] = f2bf(v.y);
    out[i + 2] = f2bf(v.z);
    out[i + 3] = f2bf(v.w);
  }
}

// ---------- C[M,N] = A[M,K] @ W[N,K]^T, bf16 in, bf16 or f32 out ----------
// Tiles fed by the Tensor Data Mover (tensor_load_to_lds) with LDS padding
// reproducing the bank-conflict-free 40-element row stride.
template <bool OUTBF>
__global__ __launch_bounds__(256) void gemm_bt_kernel(const __bf16* __restrict__ A,
                                                      const __bf16* __restrict__ W,
                                                      void* __restrict__ Cout,
                                                      int M, int N, int K) {
  constexpr int LDT = 40;  // 80B row stride: 16B aligned, conflict-free
  __shared__ __bf16 sA[2][128 * LDT];
  __shared__ __bf16 sB[2][128 * LDT];

  const int tid  = threadIdx.x;
  const int lane = tid & 31;
  const int wid  = tid >> 5;
  const int wm   = (wid & 3) * 32;   // 4 waves down
  const int wn   = (wid >> 2) * 64;  // 2 waves across
  const int m0   = blockIdx.y * 128;
  const int n0   = blockIdx.x * 128;

  v8f acc[2][4];
#pragma unroll
  for (int i = 0; i < 2; ++i)
#pragma unroll
    for (int j = 0; j < 4; ++j) acc[i][j] = (v8f){0.f, 0.f, 0.f, 0.f, 0.f, 0.f, 0.f, 0.f};

#if GEMM_USE_TDM
  // D# group1 word0: data_size=2B(code1)<<16 | pad_enable<<20 |
  //                  pad_interval=16DW(code3)<<22 | pad_amount=4DW(code3)<<25
  const unsigned g1w0 = 0x06D10000u;
  auto issue_tile = [&](int kt, int buf) {
    if (wid != 0) return;
    unsigned long long gaA =
        (unsigned long long)(size_t)A + (((size_t)m0 * K + (size_t)kt * 32) << 1);
    unsigned long long gaB =
        (unsigned long long)(size_t)W + (((size_t)n0 * K + (size_t)kt * 32) << 1);
    u32x4 a0 = {1u, lds_byte_addr(&sA[buf][0]), (unsigned)gaA,
                ((unsigned)((gaA >> 32) & 0x01FFFFFFu)) | 0x80000000u};  // type=2
    u32x4 b0 = {1u, lds_byte_addr(&sB[buf][0]), (unsigned)gaB,
                ((unsigned)((gaB >> 32) & 0x01FFFFFFu)) | 0x80000000u};
    // tensor: dim0=K, dim1=rows(M|N), stride0=K ; tile: 32 x 128
    i32x8 a1 = {(int)g1w0,
                (int)(((unsigned)K & 0xffffu) << 16),
                (int)((((unsigned)K >> 16) & 0xffffu) | (((unsigned)M & 0xffffu) << 16)),
                (int)((((unsigned)M >> 16) & 0xffffu) | (32u << 16)),
                128, K, 0, 0};
    i32x8 b1 = a1;
    b1[2] = (int)((((unsigned)K >> 16) & 0xffffu) | (((unsigned)N & 0xffffu) << 16));
    b1[3] = (int)((((unsigned)N >> 16) & 0xffffu) | (32u << 16));
    i32x4 z4 = {0, 0, 0, 0};
    i32x8 z8 = {0, 0, 0, 0, 0, 0, 0, 0};
    __builtin_amdgcn_tensor_load_to_lds(a0, a1, z4, z4, z8, 0);
    __builtin_amdgcn_tensor_load_to_lds(b0, b1, z4, z4, z8, 0);
  };
#else
  const int lr = tid >> 1;        // 0..127 : tile row loaded by this thread
  const int lc = (tid & 1) * 16;  // 0 / 16 : k sub-offset
  auto issue_tile = [&](int kt, int buf) {
    const __bf16* ga = A + (size_t)(m0 + lr) * K + kt * 32 + lc;
    uint4 a0 = *(const uint4*)ga;
    uint4 a1 = *(const uint4*)(ga + 8);
    *(uint4*)&sA[buf][lr * LDT + lc]     = a0;
    *(uint4*)&sA[buf][lr * LDT + lc + 8] = a1;
    const __bf16* gb = W + (size_t)(n0 + lr) * K + kt * 32 + lc;
    uint4 b0 = *(const uint4*)gb;
    uint4 b1 = *(const uint4*)(gb + 8);
    *(uint4*)&sB[buf][lr * LDT + lc]     = b0;
    *(uint4*)&sB[buf][lr * LDT + lc + 8] = b1;
  };
#endif

  issue_tile(0, 0);
  const int nk = K / 32;
  for (int kt = 0; kt < nk; ++kt) {
#if GEMM_USE_TDM
    __builtin_amdgcn_s_wait_tensorcnt(0);  // no-op for non-issuing waves
#endif
    __syncthreads();  // tile kt visible; previous compute fully done
    if (kt + 1 < nk) issue_tile(kt + 1, (kt + 1) & 1);
    const int buf = kt & 1;
    v16bf af[2], bfv[4];
#pragma unroll
    for (int i = 0; i < 2; ++i) af[i] = ldsA(&sA[buf][(wm + 16 * i) * LDT], LDT, lane);
#pragma unroll
    for (int j = 0; j < 4; ++j) bfv[j] = ldsB(&sB[buf][(wn + 16 * j) * LDT], LDT, lane);
#pragma unroll
    for (int i = 0; i < 2; ++i)
#pragma unroll
      for (int j = 0; j < 4; ++j) acc[i][j] = wmma_bf16(af[i], bfv[j], acc[i][j]);
  }

  // C/D layout: lane n = lane&15, VGPR r -> row r (lanes 0-15) or 8+r (16-31)
#pragma unroll
  for (int i = 0; i < 2; ++i)
#pragma unroll
    for (int j = 0; j < 4; ++j) {
      int row0 = m0 + wm + 16 * i + ((lane & 16) ? 8 : 0);
      int col  = n0 + wn + 16 * j + (lane & 15);
#pragma unroll
      for (int r = 0; r < 8; ++r) {
        size_t off = (size_t)(row0 + r) * N + col;
        float v    = acc[i][j][r];
        if (OUTBF)
          ((__bf16*)Cout)[off] = f2bf(v);
        else
          ((float*)Cout)[off] = v;
      }
    }
}

// ---------- vsum[n,h,d] = sum_k mask[k] * v3[n,k,h,d] ----------
__global__ __launch_bounds__(128) void vsum_kernel(const __bf16* __restrict__ V,
                                                   const unsigned char* __restrict__ am,
                                                   float* __restrict__ vsum) {
  int n = blockIdx.x / H_NUM;
  int h = blockIdx.x % H_NUM;
  int d = threadIdx.x;
  float acc = 0.f;
  for (int k = 0; k < NB * BL; ++k) {
    int sp = n * BL + k - BL;  // concat key k -> seq position (zero-padded ends)
    if (sp >= 0 && sp < S_DIM && am[k])
      acc += bf2f(V[(size_t)sp * E_DIM + h * D_DIM + d]);
  }
  vsum[(n * H_NUM + h) * D_DIM + d] = acc;
}

// ---------- local attention, flash-style, all-WMMA ----------
__global__ __launch_bounds__(256) void attn_kernel(const __bf16* __restrict__ Q,
                                                   const __bf16* __restrict__ Kt,
                                                   const __bf16* __restrict__ Vt,
                                                   const unsigned char* __restrict__ am,
                                                   const float* __restrict__ vsum,
                                                   __bf16* __restrict__ Out) {
  constexpr int LDK = 136;  // K tile stride (keys x d)
  constexpr int LDV = 72;   // Vt tile stride (d x keys), also P stride
  __shared__ __bf16 sK[64 * LDK];       // [key][d]
  __shared__ __bf16 sVt[D_DIM * LDV];   // [d][key] (transposed)
  __shared__ __bf16 sP[8][16 * LDV];    // per-wave P tile [q][key]

  const int tid  = threadIdx.x;
  const int lane = tid & 31;
  const int wid  = tid >> 5;
  const int n    = blockIdx.z;
  const int h    = blockIdx.y;
  const int q0   = blockIdx.x * 128;
  const int qbase = wid * 16;  // 8 waves x 16 queries

  // preload Q fragments (A layout straight from global; row stride E_DIM)
  v16bf qf[4];
  {
    int m  = lane & 15;
    int kb = (lane & 16) ? 8 : 0;
    const __bf16* qp = Q + (size_t)(n * BL + q0 + qbase + m) * E_DIM + h * D_DIM;
#pragma unroll
    for (int c = 0; c < 4; ++c) {
      FragCvt f;
      f.u.lo = *(const uint4*)(qp + 32 * c + kb);
      f.u.hi = *(const uint4*)(qp + 32 * c + kb + 16);
      qf[c]  = f.v;
    }
  }

  float qmf[8];
#pragma unroll
  for (int r = 0; r < 8; ++r) {
    int qr  = ((lane & 16) ? 8 : 0) + r;
    qmf[r]  = (float)am[n * BL + q0 + qbase + qr];
  }

  v8f oacc[8];
#pragma unroll
  for (int j = 0; j < 8; ++j) oacc[j] = (v8f){0.f, 0.f, 0.f, 0.f, 0.f, 0.f, 0.f, 0.f};
  float mrun[8], lrun[8];
#pragma unroll
  for (int r = 0; r < 8; ++r) { mrun[r] = 0.f; lrun[r] = 1.f; }  // sink exp(-m) folded in

  const int ldr = tid >> 2;         // 0..63 : key row
  const int ldc = (tid & 3) * 32;   // d sub-offset
  __bf16* pw = &sP[wid][0];

  for (int t = 0; t < (NB * BL) / 64; ++t) {
    const int kt0 = t * 64;
    // ---- stage global loads into regs (overlap with prior compute) ----
    int gk  = kt0 + ldr;
    int sp  = n * BL + gk - BL;
    bool valid = (sp >= 0 && sp < S_DIM);
    int spc = valid ? sp : 0;
    const __bf16* kp = Kt + (size_t)spc * E_DIM + h * D_DIM + ldc;
    const __bf16* vp = Vt + (size_t)spc * E_DIM + h * D_DIM + ldc;
    uint4 kreg[4], vreg[4];
    uint4 z = {0u, 0u, 0u, 0u};
#pragma unroll
    for (int q4 = 0; q4 < 4; ++q4) {
      kreg[q4] = valid ? *(const uint4*)(kp + q4 * 8) : z;
      vreg[q4] = valid ? *(const uint4*)(vp + q4 * 8) : z;
    }
    __syncthreads();  // previous tile fully consumed
#pragma unroll
    for (int q4 = 0; q4 < 4; ++q4) {
      *(uint4*)&sK[ldr * LDK + ldc + q4 * 8] = kreg[q4];
      union { uint4 u; __bf16 b[8]; } tv; tv.u = vreg[q4];
#pragma unroll
      for (int e = 0; e < 8; ++e) sVt[(ldc + q4 * 8 + e) * LDV + ldr] = tv.b[e];
    }
    __syncthreads();  // tile ready

    // ---- scores: 16 q x 64 keys via 4x4 WMMA ----
    v8f sc[4];
#pragma unroll
    for (int g = 0; g < 4; ++g) {
      sc[g] = (v8f){0.f, 0.f, 0.f, 0.f, 0.f, 0.f, 0.f, 0.f};
#pragma unroll
      for (int c = 0; c < 4; ++c)
        sc[g] = wmma_bf16(qf[c], ldsB(sK + (g * 16) * LDK + 32 * c, LDK, lane), sc[g]);
    }

    // ---- mask (+1 on allowed), online softmax with sink ----
    float kmb[4];
#pragma unroll
    for (int g = 0; g < 4; ++g) {
      int kc  = kt0 + g * 16 + (lane & 15);
      int ksp = n * BL + kc - BL;
      kmb[g]  = (ksp >= 0 && ksp < S_DIM && am[ksp]) ? 1.f : 0.f;
    }
#pragma unroll
    for (int r = 0; r < 8; ++r) {
      int qr    = ((lane & 16) ? 8 : 0) + r;
      int qglob = BL + q0 + qbase + qr;  // query position in concat coords
      float rmax = -1e30f;
#pragma unroll
      for (int g = 0; g < 4; ++g) {
        int kc  = kt0 + g * 16 + (lane & 15);
        int rel = kc - qglob;
        bool allowed = (kmb[g] != 0.f) && (qmf[r] != 0.f) && (rel <= 0) && (rel > -BL);
        float v = sc[g][r] + (allowed ? 1.f : 0.f);
        sc[g][r] = v;
        rmax = fmaxf(rmax, v);
      }
      rmax = fmaxf(rmax, __shfl_xor(rmax, 1, 32));
      rmax = fmaxf(rmax, __shfl_xor(rmax, 2, 32));
      rmax = fmaxf(rmax, __shfl_xor(rmax, 4, 32));
      rmax = fmaxf(rmax, __shfl_xor(rmax, 8, 32));
      float mnew  = fmaxf(mrun[r], rmax);
      float scale = __expf(mrun[r] - mnew);
      float rsum  = 0.f;
#pragma unroll
      for (int g = 0; g < 4; ++g) {
        float e  = __expf(sc[g][r] - mnew);
        sc[g][r] = e;
        rsum += e;
      }
      rsum += __shfl_xor(rsum, 1, 32);
      rsum += __shfl_xor(rsum, 2, 32);
      rsum += __shfl_xor(rsum, 4, 32);
      rsum += __shfl_xor(rsum, 8, 32);
      lrun[r] = lrun[r] * scale + rsum;
      mrun[r] = mnew;
#pragma unroll
      for (int j = 0; j < 8; ++j) oacc[j][r] *= scale;
    }

    // ---- P -> wave-private LDS, then PV WMMAs ----
#pragma unroll
    for (int r = 0; r < 8; ++r) {
      int qr = ((lane & 16) ? 8 : 0) + r;
#pragma unroll
      for (int g = 0; g < 4; ++g)
        pw[qr * LDV + g * 16 + (lane & 15)] = f2bf(sc[g][r]);
    }
    asm volatile("s_wait_dscnt 0" ::: "memory");
#pragma unroll
    for (int c = 0; c < 2; ++c) {
      v16bf af = ldsA(pw + 32 * c, LDV, lane);
#pragma unroll
      for (int j = 0; j < 8; ++j)
        oacc[j] = wmma_bf16(af, ldsB(sVt + (j * 16) * LDV + 32 * c, LDV, lane), oacc[j]);
    }
  }

  // ---- epilogue: out = acc/l + vsum ----
  const float* vs = vsum + (n * H_NUM + h) * D_DIM;
#pragma unroll
  for (int j = 0; j < 8; ++j) {
    int dcol = j * 16 + (lane & 15);
    float vsd = vs[dcol];
#pragma unroll
    for (int r = 0; r < 8; ++r) {
      int qr = ((lane & 16) ? 8 : 0) + r;
      int sq = n * BL + q0 + qbase + qr;
      float v = oacc[j][r] / lrun[r] + vsd;
      Out[(size_t)sq * E_DIM + h * D_DIM + dcol] = f2bf(v);
    }
  }
}

extern "C" void kernel_launch(void* const* d_in, const int* in_sizes, int n_in,
                              void* d_out, int out_size, void* d_ws, size_t ws_size,
                              hipStream_t stream) {
  const float* hs = (const float*)d_in[0];
  const unsigned char* am = (const unsigned char*)d_in[1];
  const float* wq = (const float*)d_in[2];
  const float* wk = (const float*)d_in[3];
  const float* wv = (const float*)d_in[4];
  const float* wo = (const float*)d_in[5];

  char* ws = (char*)d_ws;
  size_t off = 0;
  auto alloc = [&](size_t bytes) {
    void* p = ws + off;
    off += (bytes + 255) & ~(size_t)255;
    return p;
  };
  const size_t ne = (size_t)S_DIM * E_DIM;
  const size_t nw = (size_t)E_DIM * E_DIM;
  __bf16* Xh = (__bf16*)alloc(ne * 2);
  __bf16* Wq = (__bf16*)alloc(nw * 2);
  __bf16* Wk = (__bf16*)alloc(nw * 2);
  __bf16* Wv = (__bf16*)alloc(nw * 2);
  __bf16* Wo = (__bf16*)alloc(nw * 2);
  __bf16* Qb = (__bf16*)alloc(ne * 2);
  __bf16* Kb = (__bf16*)alloc(ne * 2);
  __bf16* Vb = (__bf16*)alloc(ne * 2);
  __bf16* Ab = (__bf16*)alloc(ne * 2);
  float*  Vs = (float*)alloc((size_t)NB * H_NUM * D_DIM * 4);

  cast_kernel<<<(int)(ne / 4 / 256), 256, 0, stream>>>(hs, Xh, (int)ne);
  cast_kernel<<<(int)(nw / 4 / 256), 256, 0, stream>>>(wq, Wq, (int)nw);
  cast_kernel<<<(int)(nw / 4 / 256), 256, 0, stream>>>(wk, Wk, (int)nw);
  cast_kernel<<<(int)(nw / 4 / 256), 256, 0, stream>>>(wv, Wv, (int)nw);
  cast_kernel<<<(int)(nw / 4 / 256), 256, 0, stream>>>(wo, Wo, (int)nw);

  dim3 gg(E_DIM / 128, S_DIM / 128);
  gemm_bt_kernel<true><<<gg, 256, 0, stream>>>(Xh, Wq, Qb, S_DIM, E_DIM, E_DIM);
  gemm_bt_kernel<true><<<gg, 256, 0, stream>>>(Xh, Wk, Kb, S_DIM, E_DIM, E_DIM);
  gemm_bt_kernel<true><<<gg, 256, 0, stream>>>(Xh, Wv, Vb, S_DIM, E_DIM, E_DIM);

  vsum_kernel<<<NB * H_NUM, D_DIM, 0, stream>>>(Vb, am, Vs);

  dim3 ag(BL / 128, H_NUM, NB);
  attn_kernel<<<ag, 256, 0, stream>>>(Qb, Kb, Vb, am, Vs, Ab);

  gemm_bt_kernel<false><<<gg, 256, 0, stream>>>(Ab, Wo, d_out, S_DIM, E_DIM, E_DIM);
}
